// NerfactoModel_6038724018410
// MI455X (gfx1250) — compile-verified, matrix operand order checked by reading, not run on hardware
//
#include <hip/hip_runtime.h>
#include <cmath>

typedef _Float16 half_t;
typedef __attribute__((ext_vector_type(16))) _Float16 v16h;
typedef __attribute__((ext_vector_type(8)))  _Float16 v8h;
typedef __attribute__((ext_vector_type(8)))  float    v8f;

struct HashTabs { const float* t[16]; int res[16]; };

// workspace layout, units of halves for weights
#define WD1T 0        // [64][32]  density L1, transposed
#define WD2T 2048     // [16][64]  density L2
#define WC1T 3072     // [64][64]  color L1 (K row 63 zero-padded)
#define WC2T 7168     // [64][64]  color L2
#define WC3T 11264    // [16][64]  color L3 (cols >=3 zero)
#define NW_HALF 12288
// bias pack (floats), after the weight pack
#define BD1 0
#define BD2 64
#define BC1 80
#define BC2 144
#define BC3 208
#define NB  224

__device__ __forceinline__ int imin(int a, int b) { return a < b ? a : b; }
__device__ __forceinline__ int imax(int a, int b) { return a > b ? a : b; }

__device__ __forceinline__ v8f wmma16(v16h a, v16h b, v8f c) {
  // D = A(16x32 f16) * B(32x16 f16) + C(16x16 f32)
  return __builtin_amdgcn_wmma_f32_16x16x32_f16(false, a, false, b, (short)0, c,
                                                false, false);
}

// Load a 16-bit A/B fragment from an LDS matrix with row stride `rowStride`
// halves. Lane half h owns K = kb+8h..kb+8h+7 and kb+16+8h..kb+16+8h+7,
// i.e. two contiguous 16-byte runs -> two ds_load_b128.
__device__ __forceinline__ v16h load_frag(const half_t* base, int row,
                                          int rowStride, int h, int kb) {
  const half_t* p = base + row * rowStride + kb + 8 * h;
  v8h lo = *(const v8h*)(p);
  v8h hi = *(const v8h*)(p + 16);
  v16h a;
#pragma unroll
  for (int i = 0; i < 8; ++i) { a[i] = lo[i]; a[i + 8] = hi[i]; }
  return a;
}

// Store a D fragment (f32) into the 16x64 f16 staging tile, optional ReLU.
// C/D layout: VGPR r -> row (r + 8h), column = lane&15 (caller passes col).
__device__ __forceinline__ void store_stage(half_t* stage, int h, int col,
                                            v8f c, bool relu) {
#pragma unroll
  for (int r = 0; r < 8; ++r) {
    float v = c[r];
    if (relu) v = fmaxf(v, 0.0f);
    stage[(r + 8 * h) * 64 + col] = (half_t)v;
  }
}

// ---------------------------------------------------------------------------
// Prep: convert all MLP weights to f16, transposed to [n][k] with K padding,
// and pack biases. Rewritten every launch (deterministic).
// ---------------------------------------------------------------------------
__global__ void __launch_bounds__(256)
nerf_prep_kernel(const float* __restrict__ dW1, const float* __restrict__ db1,
                 const float* __restrict__ dW2, const float* __restrict__ db2,
                 const float* __restrict__ cW1, const float* __restrict__ cb1,
                 const float* __restrict__ cW2, const float* __restrict__ cb2,
                 const float* __restrict__ cW3, const float* __restrict__ cb3,
                 half_t* __restrict__ wsW, float* __restrict__ wsB) {
  int idx = blockIdx.x * 256 + threadIdx.x;
  if (idx < 2048) {                       // Wd1t [64][32]: W (32x64)
    int n = idx >> 5, k = idx & 31;
    wsW[idx] = (half_t)dW1[k * 64 + n];
  } else if (idx < 3072) {                // Wd2t [16][64]: W (64x16)
    int i = idx - WD2T; int n = i >> 6, k = i & 63;
    wsW[idx] = (half_t)dW2[k * 16 + n];
  } else if (idx < 7168) {                // Wc1t [64][64]: W (63x64), k=63 -> 0
    int i = idx - WC1T; int n = i >> 6, k = i & 63;
    wsW[idx] = (k < 63) ? (half_t)cW1[k * 64 + n] : (half_t)0.0f;
  } else if (idx < 11264) {               // Wc2t [64][64]: W (64x64)
    int i = idx - WC2T; int n = i >> 6, k = i & 63;
    wsW[idx] = (half_t)cW2[k * 64 + n];
  } else if (idx < NW_HALF) {             // Wc3t [16][64]: W (64x3), n>=3 -> 0
    int i = idx - WC3T; int n = i >> 6, k = i & 63;
    wsW[idx] = (n < 3) ? (half_t)cW3[k * 3 + n] : (half_t)0.0f;
  } else if (idx < NW_HALF + NB) {        // biases
    int j = idx - NW_HALF;
    float v;
    if (j < 64)       v = db1[j];
    else if (j < 80)  v = db2[j - 64];
    else if (j < 144) v = cb1[j - 80];
    else if (j < 208) v = cb2[j - 144];
    else              v = (j - 208 < 3) ? cb3[j - 208] : 0.0f;
    wsB[j] = v;
  }
}

// ---------------------------------------------------------------------------
// Fused field forward: hash encode + SH encode + density MLP + color MLP.
// One wave = one 16-sample tile; 24 WMMAs per tile.
// ---------------------------------------------------------------------------
__global__ void __launch_bounds__(256)
nerf_field_kernel(const float* __restrict__ pos, const float* __restrict__ dirs,
                  const int* __restrict__ cam, const float* __restrict__ aabb,
                  const float* __restrict__ app, HashTabs tabs,
                  const half_t* __restrict__ wsW, const float* __restrict__ wsB,
                  float* __restrict__ out, int nSamples, int numTiles) {
  __shared__ __align__(16) half_t sW[NW_HALF];      // f16 weight pack (24 KB)
  __shared__ float sB[NB];                          // bias pack
  __shared__ __align__(16) half_t sStage[8][16 * 64]; // per-wave 16x64 tiles
  __shared__ const float* sTab[16];
  __shared__ int sRes[16];

  const int tid = threadIdx.x;
  {
    const uint32_t* src = (const uint32_t*)wsW;
    uint32_t* dst = (uint32_t*)sW;
    for (int i = tid; i < NW_HALF / 2; i += 256) dst[i] = src[i];
    for (int i = tid; i < NB; i += 256) sB[i] = wsB[i];
    if (tid < 16) { sTab[tid] = tabs.t[tid]; sRes[tid] = tabs.res[tid]; }
  }
  __syncthreads();

  const int wave = tid >> 5;
  const int lane = tid & 31;
  const int m = lane & 15;     // sample-in-tile / output column
  const int h = lane >> 4;     // lane half (K/row split per WMMA layouts)
  half_t* stage = &sStage[wave][0];

  const float lo0 = aabb[0], lo1 = aabb[1], lo2 = aabb[2];
  const float iv0 = 1.0f / (aabb[3] - lo0);
  const float iv1 = 1.0f / (aabb[4] - lo1);
  const float iv2 = 1.0f / (aabb[5] - lo2);

  float* outRgb = out;
  float* outDen = out + 3 * (size_t)nSamples;

  for (int tile = blockIdx.x * 8 + wave; tile < numTiles; tile += gridDim.x * 8) {
    const int gs = tile * 16 + m;

    // ---- normalized position ----
    float px = fminf(fmaxf((pos[gs * 3 + 0] - lo0) * iv0, 0.0f), 1.0f);
    float py = fminf(fmaxf((pos[gs * 3 + 1] - lo1) * iv1, 0.0f), 1.0f);
    float pz = fminf(fmaxf((pos[gs * 3 + 2] - lo2) * iv2, 0.0f), 1.0f);

    // ---- hash encode, built directly in WMMA A-fragment layout ----
    // lane half h owns K = {8h..8h+7, 16+8h..16+8h+7} = levels
    // {4h..4h+3, 8+4h..8+4h+3}, two f16 features per level.
    v16h aFeat;
#pragma unroll
    for (int j = 0; j < 8; ++j) {
      int level = (j < 4) ? (4 * h + j) : (4 + 4 * h + j);
      int rm = sRes[level] - 1;
      int cx = imin(imax((int)floorf(px * rm), 0), rm);
      int cy = imin(imax((int)floorf(py * rm), 0), rm);
      int cz = imin(imax((int)floorf(pz * rm), 0), rm);
      int hidx = (cx + cy * 2481 + cz * 1941) & 4095; // primes mod 4096
      const float2 f = *(const float2*)(sTab[level] + 2 * hidx);
      aFeat[2 * j]     = (half_t)f.x;
      aFeat[2 * j + 1] = (half_t)f.y;
    }

    // ---- density MLP layer 1: (16x32) @ (32x64), bias in C, ReLU ----
#pragma unroll
    for (int t = 0; t < 4; ++t) {
      v16h b = load_frag(sW + WD1T, t * 16 + m, 32, h, 0);
      v8f c; float bv = sB[BD1 + t * 16 + m];
#pragma unroll
      for (int r = 0; r < 8; ++r) c[r] = bv;
      c = wmma16(aFeat, b, c);
      store_stage(stage, h, t * 16 + m, c, true);
    }

    // ---- density MLP layer 2: (16x64) @ (64x16) ----
    {
      v16h a0 = load_frag(stage, m, 64, h, 0);
      v16h a1 = load_frag(stage, m, 64, h, 32);
      v16h b0 = load_frag(sW + WD2T, m, 64, h, 0);
      v16h b1 = load_frag(sW + WD2T, m, 64, h, 32);
      v8f c; float bv = sB[BD2 + m];
#pragma unroll
      for (int r = 0; r < 8; ++r) c[r] = bv;
      c = wmma16(a0, b0, c);
      c = wmma16(a1, b1, c);
      if (m == 0) {          // col 0 -> density (ReLU)
#pragma unroll
        for (int r = 0; r < 8; ++r)
          outDen[tile * 16 + r + 8 * h] = fmaxf(c[r], 0.0f);
      } else {               // cols 1..15 -> geo features into cin cols 0..14
#pragma unroll
        for (int r = 0; r < 8; ++r)
          stage[(r + 8 * h) * 64 + (m - 1)] = (half_t)c[r];
      }
    }

    // ---- SH encode of direction -> cin cols 15..30 ----
    {
      float x = dirs[gs * 3 + 0], y = dirs[gs * 3 + 1], z = dirs[gs * 3 + 2];
      float xx = x * x, yy = y * y, zz = z * z;
      float sh[16];
      sh[0]  = 0.28209479177387814f;
      sh[1]  = 0.4886025119029199f * y;
      sh[2]  = 0.4886025119029199f * z;
      sh[3]  = 0.4886025119029199f * x;
      sh[4]  = 1.0925484305920792f * x * y;
      sh[5]  = 1.0925484305920792f * y * z;
      sh[6]  = 0.9461746957575601f * (2.0f * zz - xx - yy);
      sh[7]  = 1.0925484305920792f * z * x;
      sh[8]  = 0.5462742152960396f * (xx - yy);
      sh[9]  = 0.5900435899266435f * y * (3.0f * xx - yy);
      sh[10] = 2.890611442640554f * x * y * z;
      sh[11] = 0.4570457994644658f * y * (5.0f * zz - 1.0f);
      sh[12] = 0.3731763325901154f * z * (5.0f * zz - 3.0f);
      sh[13] = 0.4570457994644658f * x * (5.0f * zz - 1.0f);
      sh[14] = 1.445305721320277f * z * (xx - yy);
      sh[15] = 0.5900435899266435f * x * (xx - 3.0f * yy);
#pragma unroll
      for (int j = 0; j < 8; ++j) {
        float bv = (h == 0) ? sh[j] : sh[8 + j]; // keep sh[] in registers
        stage[m * 64 + 15 + 8 * h + j] = (half_t)bv;
      }
    }

    // ---- appearance embedding -> cin cols 31..62, pad col 63 ----
    {
      int ci = cam[gs];
      const float4* ap = (const float4*)(app + (size_t)ci * 32 + 16 * h);
#pragma unroll
      for (int q = 0; q < 4; ++q) {
        float4 v = ap[q];
        int c0 = m * 64 + 31 + 16 * h + 4 * q;
        stage[c0 + 0] = (half_t)v.x;
        stage[c0 + 1] = (half_t)v.y;
        stage[c0 + 2] = (half_t)v.z;
        stage[c0 + 3] = (half_t)v.w;
      }
      stage[m * 64 + 63] = (half_t)0.0f;
    }

    // ---- color MLP: 64(padded)->64 relu -> 64 relu -> 16(padded) ----
    {
      v16h a0 = load_frag(stage, m, 64, h, 0);
      v16h a1 = load_frag(stage, m, 64, h, 32);
#pragma unroll
      for (int t = 0; t < 4; ++t) {
        v16h b0 = load_frag(sW + WC1T, t * 16 + m, 64, h, 0);
        v16h b1 = load_frag(sW + WC1T, t * 16 + m, 64, h, 32);
        v8f c; float bv = sB[BC1 + t * 16 + m];
#pragma unroll
        for (int r = 0; r < 8; ++r) c[r] = bv;
        c = wmma16(a0, b0, c);
        c = wmma16(a1, b1, c);
        store_stage(stage, h, t * 16 + m, c, true);
      }
      a0 = load_frag(stage, m, 64, h, 0);
      a1 = load_frag(stage, m, 64, h, 32);
#pragma unroll
      for (int t = 0; t < 4; ++t) {
        v16h b0 = load_frag(sW + WC2T, t * 16 + m, 64, h, 0);
        v16h b1 = load_frag(sW + WC2T, t * 16 + m, 64, h, 32);
        v8f c; float bv = sB[BC2 + t * 16 + m];
#pragma unroll
        for (int r = 0; r < 8; ++r) c[r] = bv;
        c = wmma16(a0, b0, c);
        c = wmma16(a1, b1, c);
        store_stage(stage, h, t * 16 + m, c, true);
      }
      a0 = load_frag(stage, m, 64, h, 0);
      a1 = load_frag(stage, m, 64, h, 32);
      v16h b0 = load_frag(sW + WC3T, m, 64, h, 0);
      v16h b1 = load_frag(sW + WC3T, m, 64, h, 32);
      v8f c; float bv = sB[BC3 + m];
#pragma unroll
      for (int r = 0; r < 8; ++r) c[r] = bv;
      c = wmma16(a0, b0, c);
      c = wmma16(a1, b1, c);
      if (m < 3) {           // cols 0..2 -> sigmoid -> rgb
#pragma unroll
        for (int r = 0; r < 8; ++r) {
          int row = tile * 16 + r + 8 * h;
          outRgb[(size_t)row * 3 + m] = 1.0f / (1.0f + expf(-c[r]));
        }
      }
    }
  }
}

// ---------------------------------------------------------------------------
// d_in order: 0 ray_samples, 1 directions, 2 camera_indices, 3 aabb,
// 4..19 hash_tables[0..15], 20 appearance_table,
// 21 dW1, 22 db1, 23 dW2, 24 db2, 25 cW1, 26 cb1, 27 cW2, 28 cb2, 29 cW3, 30 cb3
// d_out: rgb (N*3 floats) then density (N floats)
// ---------------------------------------------------------------------------
extern "C" void kernel_launch(void* const* d_in, const int* in_sizes, int n_in,
                              void* d_out, int out_size, void* d_ws, size_t ws_size,
                              hipStream_t stream) {
  (void)n_in; (void)out_size; (void)ws_size;
  const float* pos  = (const float*)d_in[0];
  const float* dirs = (const float*)d_in[1];
  const int*   cam  = (const int*)d_in[2];
  const float* aabb = (const float*)d_in[3];

  HashTabs tabs;
  const double g = exp(log(2048.0 / 16.0) / 15.0);
  for (int i = 0; i < 16; ++i) {
    tabs.t[i] = (const float*)d_in[4 + i];
    tabs.res[i] = (int)(16.0 * pow(g, (double)i)); // matches int(MIN_RES*GROWTH**i)
  }
  const float* app = (const float*)d_in[20];

  half_t* wsW = (half_t*)d_ws;
  float*  wsB = (float*)((char*)d_ws + NW_HALF * sizeof(half_t));

  nerf_prep_kernel<<<49, 256, 0, stream>>>(
      (const float*)d_in[21], (const float*)d_in[22],
      (const float*)d_in[23], (const float*)d_in[24],
      (const float*)d_in[25], (const float*)d_in[26],
      (const float*)d_in[27], (const float*)d_in[28],
      (const float*)d_in[29], (const float*)d_in[30], wsW, wsB);

  const int nSamples = in_sizes[0] / 3;   // 786432
  const int numTiles = nSamples / 16;     // 49152
  int blocks = (numTiles + 7) / 8;
  if (blocks > 6144) blocks = 6144;

  nerf_field_kernel<<<blocks, 256, 0, stream>>>(
      pos, dirs, cam, aabb, app, tabs, wsW, wsB,
      (float*)d_out, nSamples, numTiles);
}
